// Model_15796889715486
// MI455X (gfx1250) — matrix operand order repaired and checked by reading, not compiled, not run on hardware
//
#include <hip/hip_runtime.h>
#include <hip/hip_bf16.h>

// ---------------------------------------------------------------------------
// ABWIM on gfx1250: all GEMMs via v_wmma_f32_16x16x32_f16 (f32 accumulate).
// Dims: B=64, Sq=40, Sw=12, Sr=4, E=300(->320), H=256, 2H=512, 4H=1024, NF=150(->160)
// ---------------------------------------------------------------------------

typedef _Float16 f16;
typedef __attribute__((ext_vector_type(16))) _Float16 v16h;
typedef __attribute__((ext_vector_type(8)))  _Float16 v8h;
typedef __attribute__((ext_vector_type(8)))  float    v8f;

#define DEV __device__ __forceinline__

// A-matrix 16x32 f16 fragment (row-major A, lda in elements, multiple of 8).
// lane L: m = L&15; k = (e<8 ? e : e+8) + (L>>4)*8  -> two contiguous v8h loads.
DEV v16h load_frag_a(const f16* p, int ld) {
  int l = threadIdx.x & 31;
  const f16* r = p + (size_t)(l & 15) * ld + ((l >> 4) << 3);
  v8h lo = *(const v8h*)r;
  v8h hi = *(const v8h*)(r + 16);
  v16h a;
#pragma unroll
  for (int i = 0; i < 8; ++i) { a[i] = lo[i]; a[i + 8] = hi[i]; }
  return a;
}

// B-matrix 32x16 f16 fragment from row-major (N,K) operand ("TN" GEMM).
// lane L: n = L&15; k = (e<8 ? e : e+8) + (L>>4)*8  -> two contiguous v8h loads.
DEV v16h load_frag_b(const f16* p, int ld) {
  int l = threadIdx.x & 31;
  const f16* r = p + (size_t)(l & 15) * ld + ((l >> 4) << 3);
  v8h lo = *(const v8h*)r;
  v8h hi = *(const v8h*)(r + 16);
  v16h b;
#pragma unroll
  for (int i = 0; i < 8; ++i) { b[i] = lo[i]; b[i + 8] = hi[i]; }
  return b;
}

// One wave computes a 16x16 f32 tile: D += A[16xK] * B[NxK]^T, K multiple of 32.
DEV v8f gemm_tile(const f16* A, int lda, const f16* B, int ldb, int K, v8f acc) {
  for (int k = 0; k < K; k += 32) {
    v16h a = load_frag_a(A + k, lda);
    v16h b = load_frag_b(B + k, ldb);
    acc = __builtin_amdgcn_wmma_f32_16x16x32_f16(false, a, false, b,
                                                 (short)0, acc, false, false);
  }
  return acc;
}

// Fused 4-gate tile GEMM for the LSTM: one A fragment per k-step feeds four
// WMMAs (gate weight rows are `bstride` elements apart). unroll 2 keeps the
// live set at ~2 A frags + 8 B frags + 4 accumulators -> no scratch spills.
DEV void gemm_tile4(const f16* A, int lda, const f16* B, size_t bstride,
                    int ldb, int K, v8f z[4]) {
#pragma unroll 2
  for (int k = 0; k < K; k += 32) {
    v16h a = load_frag_a(A + k, lda);
#pragma unroll
    for (int g = 0; g < 4; ++g) {
      v16h b = load_frag_b(B + (size_t)g * bstride + k, ldb);
      z[g] = __builtin_amdgcn_wmma_f32_16x16x32_f16(false, a, false, b,
                                                    (short)0, z[g], false, false);
    }
  }
}

DEV float sigm(float x)  { return 1.0f / (1.0f + __expf(-x)); }
DEV float tanh_(float x) { return 2.0f * sigm(2.0f * x) - 1.0f; }

// ------------------------------ prep kernels -------------------------------

__global__ void k_pad_f16(const float* __restrict__ src, f16* __restrict__ dst,
                          int R, int C, int Cp, int total) {
  int i = blockIdx.x * blockDim.x + threadIdx.x;
  if (i >= total) return;
  int c = i % Cp, r = i / Cp;
  dst[i] = (r < R && c < C) ? (f16)src[(size_t)r * C + c] : (f16)0.0f;
}

__global__ void k_transpose_f16(const float* __restrict__ src, f16* __restrict__ dst,
                                int R, int C) {
  int i = blockIdx.x * blockDim.x + threadIdx.x;
  if (i >= R * C) return;
  int r = i / C, c = i % C;
  dst[(size_t)c * R + r] = (f16)src[i];
}

// cnn w (O,C,K) -> dst (Op, K*C) with dst[o][k*C+c] = w[o][c][k], zero-pad o>=O
__global__ void k_cnn_reorder(const float* __restrict__ w, f16* __restrict__ dst,
                              int O, int C, int K, int total) {
  int i = blockIdx.x * blockDim.x + threadIdx.x;
  if (i >= total) return;
  int kc = K * C;
  int o = i / kc, rem = i % kc, k = rem / C, c = rem % C;
  dst[i] = (o < O) ? (f16)w[((size_t)o * C + c) * K + k] : (f16)0.0f;
}

// idx (B,S) -> x (S, B, 320) f16, zero-pad e>=E
__global__ void k_gather(const int* __restrict__ idx, const float* __restrict__ emb,
                         int E, int S, f16* __restrict__ out, int total) {
  int i = blockIdx.x * blockDim.x + threadIdx.x;
  if (i >= total) return;
  int e = i % 320, sb = i / 320, b = sb % 64, s = sb / 64;
  int row = idx[b * S + s];
  out[i] = (e < E) ? (f16)emb[(size_t)row * E + e] : (f16)0.0f;
}

// ------------------------------ LSTM kernel --------------------------------
// One workgroup (512 thr = 16 waves). Each wave owns 4 position tiles
// (16 b-rows x 16 u-cols); c stays in registers, h shared via LDS.

__global__ __launch_bounds__(512)
void k_lstm(const f16* __restrict__ x, int T, int dir,
            const f16* __restrict__ Wih, const f16* __restrict__ Whh,
            const float* __restrict__ bias,
            const f16* h0, const float* c0,
            f16* hOut, float* cOut,
            f16* __restrict__ out, int out_off, int out_bstride)
{
  __shared__ __align__(16) f16 h_s[64 * 256];
  int tid = threadIdx.x;
  int wave = tid >> 5, lane = tid & 31;
  int n = lane & 15, half8 = (lane >> 4) << 3;

  for (int i = tid; i < 64 * 256; i += 512)
    h_s[i] = h0 ? h0[i] : (f16)0.0f;

  float c_reg[4][8];
  float hn[4][8];
#pragma unroll
  for (int pt = 0; pt < 4; ++pt) {
    int g = wave * 4 + pt, b0 = (g >> 4) << 4, u0 = (g & 15) << 4;
    int u = u0 + n;
#pragma unroll
    for (int r = 0; r < 8; ++r) {
      int b = b0 + half8 + r;
      c_reg[pt][r] = c0 ? c0[b * 256 + u] : 0.0f;
    }
  }
  __syncthreads();

  for (int step = 0; step < T; ++step) {
    int t = dir ? (T - 1 - step) : step;
    const f16* xt = x + (size_t)t * 64 * 320;

    // Prefetch next timestep's x tile (64x320 f16 = 40KB) into cache while
    // this step's WMMAs run. 512 threads x 80B covers the tile.
    if (step + 1 < T) {
      int t2 = dir ? (T - 2 - step) : (step + 1);
      const char* xn = (const char*)(x + (size_t)t2 * 64 * 320) + (size_t)tid * 80;
      __builtin_prefetch(xn, 0, 0);
    }

#pragma unroll 1
    for (int pt = 0; pt < 4; ++pt) {
      int g = wave * 4 + pt, b0 = (g >> 4) << 4, u0 = (g & 15) << 4;
      v8f z[4] = {};
      const f16* xa = xt + (size_t)b0 * 320;
      gemm_tile4(xa, 320, Wih + (size_t)u0 * 320, (size_t)256 * 320, 320, 320, z);
      const f16* ha = h_s + (size_t)b0 * 256;
      gemm_tile4(ha, 256, Whh + (size_t)u0 * 256, (size_t)256 * 256, 256, 256, z);
      int u = u0 + n;
      float bi = bias[u], bf = bias[256 + u], bg = bias[512 + u], bo = bias[768 + u];
#pragma unroll
      for (int r = 0; r < 8; ++r) {
        float iv = sigm(z[0][r] + bi);
        float fv = sigm(z[1][r] + bf);
        float gv = tanh_(z[2][r] + bg);
        float ov = sigm(z[3][r] + bo);
        float c  = fv * c_reg[pt][r] + iv * gv;
        c_reg[pt][r] = c;
        hn[pt][r] = ov * tanh_(c);
      }
    }
    __syncthreads();   // all h reads done before overwrite
#pragma unroll 1
    for (int pt = 0; pt < 4; ++pt) {
      int g = wave * 4 + pt, b0 = (g >> 4) << 4, u0 = (g & 15) << 4;
      int u = u0 + n;
#pragma unroll
      for (int r = 0; r < 8; ++r) {
        int b = b0 + half8 + r;
        f16 hv = (f16)hn[pt][r];
        h_s[b * 256 + u] = hv;
        out[(size_t)b * out_bstride + (size_t)t * 512 + out_off + u] = hv;
      }
    }
    __syncthreads();
  }

#pragma unroll 1
  for (int pt = 0; pt < 4; ++pt) {
    int g = wave * 4 + pt, b0 = (g >> 4) << 4, u0 = (g & 15) << 4;
    int u = u0 + n;
#pragma unroll
    for (int r = 0; r < 8; ++r) {
      int b = b0 + half8 + r;
      hOut[b * 256 + u] = h_s[b * 256 + u];
      cOut[b * 256 + u] = c_reg[pt][r];
    }
  }
}

// relation (B,16,512): rows 0..3 = rela_out, 4..15 = word_out
__global__ void k_relation(const f16* __restrict__ rela_out,
                           const f16* __restrict__ word_out,
                           f16* __restrict__ rel) {
  int i = blockIdx.x * blockDim.x + threadIdx.x;
  if (i >= 64 * 16 * 512) return;
  int h = i & 511, r = (i >> 9) & 15, b = i >> 13;
  rel[i] = (r < 4) ? rela_out[((size_t)b * 4 + r) * 512 + h]
                   : word_out[((size_t)b * 12 + (r - 4)) * 512 + h];
}

// ---------------------------- attention kernel -----------------------------
// One workgroup per batch. RW = rel@W (WMMA, LDS-staged), energy = RW@Qt^T
// (WMMA), fused softmax (scale 1/sqrt(16)), then atten[s][h] = sum_r a*rel.

__global__ __launch_bounds__(512)
void k_attention(const f16* __restrict__ rel, const f16* __restrict__ Qt,
                 const f16* __restrict__ WT, f16* __restrict__ atten)
{
  __shared__ __align__(16) f16 RW[16 * 512];
  __shared__ float en[16 * 48];
  __shared__ float red[40];
  int b = blockIdx.x;
  int tid = threadIdx.x, wave = tid >> 5, lane = tid & 31;
  int n = lane & 15, half8 = (lane >> 4) << 3;
  const f16* relb = rel + (size_t)b * 16 * 512;

  for (int tI = wave; tI < 32; tI += 16) {
    int n0 = tI << 4;
    v8f acc = {};
    acc = gemm_tile(relb, 512, WT + (size_t)n0 * 512, 512, 512, acc);
#pragma unroll
    for (int r = 0; r < 8; ++r)
      RW[(half8 + r) * 512 + n0 + n] = (f16)acc[r];
  }
  __syncthreads();

  if (wave < 3) {
    int s0 = wave << 4;
    v8f acc = {};
    acc = gemm_tile(RW, 512, Qt + (size_t)b * 40 * 512 + (size_t)s0 * 512, 512, 512, acc);
#pragma unroll
    for (int r = 0; r < 8; ++r)
      en[(half8 + r) * 48 + s0 + n] = acc[r] * 0.25f;   // 1/sqrt(R=16)
  }
  __syncthreads();

  float m = -1e30f;
  for (int i = tid; i < 640; i += 512) m = fmaxf(m, en[(i / 40) * 48 + (i % 40)]);
  for (int o = 16; o; o >>= 1) m = fmaxf(m, __shfl_xor(m, o, 32));
  if (lane == 0) red[wave] = m;
  __syncthreads();
  if (tid == 0) { float mm = red[0]; for (int i = 1; i < 16; ++i) mm = fmaxf(mm, red[i]); red[32] = mm; }
  __syncthreads();
  float mx = red[32];
  float s = 0.0f;
  for (int i = tid; i < 640; i += 512) {
    int idx = (i / 40) * 48 + (i % 40);
    float e = __expf(en[idx] - mx);
    en[idx] = e;
    s += e;
  }
  for (int o = 16; o; o >>= 1) s += __shfl_xor(s, o, 32);
  if (lane == 0) red[wave] = s;
  __syncthreads();
  if (tid == 0) { float ss = 0.0f; for (int i = 0; i < 16; ++i) ss += red[i]; red[33] = ss; }
  __syncthreads();
  float inv = 1.0f / red[33];

  for (int i = tid; i < 40 * 512; i += 512) {
    int sIdx = i >> 9, h = i & 511;
    float acc = 0.0f;
#pragma unroll
    for (int r = 0; r < 16; ++r)
      acc += en[r * 48 + sIdx] * (float)relb[r * 512 + h];
    atten[(size_t)b * 40 * 512 + i] = (f16)(acc * inv);
  }
}

__global__ void k_sub_f16(const f16* __restrict__ a, const f16* __restrict__ b_,
                          f16* __restrict__ d, int total) {
  int i = blockIdx.x * blockDim.x + threadIdx.x;
  if (i < total) d[i] = a[i] - b_[i];
}

// Generic TN GEMM: one wave per 16x16 tile, optional per-column bias, f16 out.
__global__ __launch_bounds__(256)
void k_gemm_tn(const f16* __restrict__ A, int lda,
               const f16* __restrict__ B, int ldb, int K,
               const float* bias,
               f16* __restrict__ D, int ldd, int Mtiles, int Ntiles)
{
  int wid = blockIdx.x * 8 + (threadIdx.x >> 5);
  if (wid >= Mtiles * Ntiles) return;
  int mt = wid / Ntiles, nt = wid % Ntiles;
  int lane = threadIdx.x & 31, n = lane & 15, half8 = (lane >> 4) << 3;
  v8f acc = {};
  acc = gemm_tile(A + (size_t)mt * 16 * lda, lda, B + (size_t)nt * 16 * ldb, ldb, K, acc);
  float bv = bias ? bias[nt * 16 + n] : 0.0f;
#pragma unroll
  for (int r = 0; r < 8; ++r)
    D[(size_t)(mt * 16 + half8 + r) * ldd + nt * 16 + n] = (f16)(acc[r] + bv);
}

// M (B,40,1024): [..,0:512]=Qt2, [..,512:1024]=atten_cur
__global__ void k_concat_M(const f16* __restrict__ q, const f16* __restrict__ att,
                           f16* __restrict__ Mb, int total) {
  int i = blockIdx.x * blockDim.x + threadIdx.x;
  if (i >= total) return;
  int c = i & 1023; size_t bs = (size_t)(i >> 10);
  Mb[i] = (c < 512) ? q[bs * 512 + c] : att[bs * 512 + (c - 512)];
}

// ------------------------- conv + max + score ------------------------------
// One workgroup per batch: 3 convs as TN GEMMs (im2col = contiguous rows of M),
// relu+max via LDS uint-punned atomicMax (values >= 0), then dot with lin_w.

__global__ __launch_bounds__(512)
void k_conv_score(const f16* __restrict__ Mbuf,
                  const f16* __restrict__ wr1, const f16* __restrict__ wr2,
                  const f16* __restrict__ wr3,
                  const float* __restrict__ cb1, const float* __restrict__ cb2,
                  const float* __restrict__ cb3,
                  const float* __restrict__ lin_w, float* __restrict__ outp)
{
  __shared__ float hmax[160];
  __shared__ float sred[16];
  int b = blockIdx.x;
  int tid = threadIdx.x, wave = tid >> 5, lane = tid & 31;
  int n = lane & 15, half8 = (lane >> 4) << 3;
  for (int i = tid; i < 160; i += 512) hmax[i] = 0.0f;
  __syncthreads();
  const f16* Mb = Mbuf + (size_t)b * 40 * 1024;

  for (int cfg = 0; cfg < 3; ++cfg) {
    const f16* w = (cfg == 0) ? wr1 : (cfg == 1) ? wr2 : wr3;
    const float* cb = (cfg == 0) ? cb1 : (cfg == 1) ? cb2 : cb3;
    int KK = (cfg == 0) ? 1024 : (cfg == 1) ? 3072 : 5120;
    int L  = (cfg == 0) ? 40 : (cfg == 1) ? 38 : 36;
    for (int tI = wave; tI < 30; tI += 16) {       // 10 o-tiles x 3 l-tiles
      int mt = tI / 3, nt = tI % 3;
      v8f acc = {};
      acc = gemm_tile(w + (size_t)mt * 16 * KK, KK,
                      Mb + (size_t)nt * 16 * 1024, 1024, KK, acc);
      int l = nt * 16 + n;
      if (l < L) {
#pragma unroll
        for (int r = 0; r < 8; ++r) {
          int o = mt * 16 + half8 + r;
          if (o < 150) {
            float v = fmaxf(acc[r] + cb[o], 0.0f);
            atomicMax((unsigned int*)&hmax[o], __float_as_uint(v));
          }
        }
      }
    }
  }
  __syncthreads();
  float p = 0.0f;
  for (int i = tid; i < 150; i += 512) p += hmax[i] * lin_w[i];
  for (int o = 16; o; o >>= 1) p += __shfl_xor(p, o, 32);
  if (lane == 0) sred[wave] = p;
  __syncthreads();
  if (tid == 0) { float ss = 0.0f; for (int i = 0; i < 16; ++i) ss += sred[i]; outp[b] = ss; }
}

// ------------------------------- launcher ----------------------------------

extern "C" void kernel_launch(void* const* d_in, const int* in_sizes, int n_in,
                              void* d_out, int out_size, void* d_ws, size_t ws_size,
                              hipStream_t stream) {
  (void)in_sizes; (void)n_in; (void)out_size; (void)ws_size;
  const int*   question  = (const int*)d_in[0];
  const int*   word_rel  = (const int*)d_in[1];
  const int*   rela_rel  = (const int*)d_in[2];
  const int*   word_prev = (const int*)d_in[3];
  const int*   rela_prev = (const int*)d_in[4];
  const float* word_emb  = (const float*)d_in[5];
  const float* rela_emb  = (const float*)d_in[6];
  const float* Wih_f = (const float*)d_in[7];
  const float* Whh_f = (const float*)d_in[8];
  const float* b_f   = (const float*)d_in[9];
  const float* Wih_b = (const float*)d_in[10];
  const float* Whh_b = (const float*)d_in[11];
  const float* b_b   = (const float*)d_in[12];
  const float* Wmat  = (const float*)d_in[13];
  const float* cnn1_w = (const float*)d_in[14];
  const float* cnn1_b = (const float*)d_in[15];
  const float* cnn2_w = (const float*)d_in[16];
  const float* cnn2_b = (const float*)d_in[17];
  const float* cnn3_w = (const float*)d_in[18];
  const float* cnn3_b = (const float*)d_in[19];
  const float* lin_w  = (const float*)d_in[20];
  const float* lin2_w = (const float*)d_in[21];
  const float* lin2_b = (const float*)d_in[22];
  float* outp = (float*)d_out;

  char* ws = (char*)d_ws;
  size_t off = 0;
  auto carveH = [&](size_t elems) -> f16* {
    f16* p = (f16*)(ws + off);
    off += ((elems * sizeof(f16)) + 255) & ~(size_t)255;
    return p;
  };
  auto carveF = [&](size_t elems) -> float* {
    float* p = (float*)(ws + off);
    off += ((elems * sizeof(float)) + 255) & ~(size_t)255;
    return p;
  };

  f16* wihF = carveH(1024 * 320);
  f16* wihB = carveH(1024 * 320);
  f16* whhF = carveH(1024 * 256);
  f16* whhB = carveH(1024 * 256);
  f16* WT    = carveH(512 * 512);
  f16* lin2h = carveH(512 * 512);
  f16* wr1 = carveH(160 * 1024);
  f16* wr2 = carveH(160 * 3072);
  f16* wr3 = carveH(160 * 5120);
  f16* xq  = carveH(40 * 64 * 320);
  f16* xwp = carveH(12 * 64 * 320);
  f16* xrp = carveH(4 * 64 * 320);
  f16* xwc = carveH(12 * 64 * 320);
  f16* xrc = carveH(4 * 64 * 320);
  f16* qout = carveH(64 * 40 * 512 + 16 * 512);   // slack for N-padded frag reads
  f16* wpo  = carveH(64 * 12 * 512);
  f16* rpo  = carveH(64 * 4 * 512);
  f16* wco  = carveH(64 * 12 * 512);
  f16* rco  = carveH(64 * 4 * 512);
  f16* relP = carveH(64 * 16 * 512);
  f16* relC = carveH(64 * 16 * 512);
  f16* attP = carveH(64 * 40 * 512);
  f16* attC = carveH(64 * 40 * 512);
  f16* subT = carveH(64 * 40 * 512);
  f16* qt2  = carveH(64 * 40 * 512 + 16 * 512);
  f16* Mbuf = carveH(64 * 40 * 1024 + 16 * 1024);
  f16* hstF = carveH(64 * 256);
  f16* hstB = carveH(64 * 256);
  f16* hstD = carveH(64 * 256);
  float* cstF = carveF(64 * 256);
  float* cstB = carveF(64 * 256);
  float* cstD = carveF(64 * 256);

  auto nb = [](int total) { return (total + 255) / 256; };

  // ---- weight prep ----
  k_pad_f16<<<nb(1024 * 320), 256, 0, stream>>>(Wih_f, wihF, 1024, 300, 320, 1024 * 320);
  k_pad_f16<<<nb(1024 * 320), 256, 0, stream>>>(Wih_b, wihB, 1024, 300, 320, 1024 * 320);
  k_pad_f16<<<nb(1024 * 256), 256, 0, stream>>>(Whh_f, whhF, 1024, 256, 256, 1024 * 256);
  k_pad_f16<<<nb(1024 * 256), 256, 0, stream>>>(Whh_b, whhB, 1024, 256, 256, 1024 * 256);
  k_transpose_f16<<<nb(512 * 512), 256, 0, stream>>>(Wmat, WT, 512, 512);
  k_pad_f16<<<nb(512 * 512), 256, 0, stream>>>(lin2_w, lin2h, 512, 512, 512, 512 * 512);
  k_cnn_reorder<<<nb(160 * 1024), 256, 0, stream>>>(cnn1_w, wr1, 150, 1024, 1, 160 * 1024);
  k_cnn_reorder<<<nb(160 * 3072), 256, 0, stream>>>(cnn2_w, wr2, 150, 1024, 3, 160 * 3072);
  k_cnn_reorder<<<nb(160 * 5120), 256, 0, stream>>>(cnn3_w, wr3, 150, 1024, 5, 160 * 5120);

  // ---- embedding gathers (S,B,320) ----
  k_gather<<<nb(40 * 64 * 320), 256, 0, stream>>>(question,  word_emb, 300, 40, xq,  40 * 64 * 320);
  k_gather<<<nb(12 * 64 * 320), 256, 0, stream>>>(word_prev, word_emb, 300, 12, xwp, 12 * 64 * 320);
  k_gather<<<nb(4  * 64 * 320), 256, 0, stream>>>(rela_prev, rela_emb, 300, 4,  xrp, 4 * 64 * 320);
  k_gather<<<nb(12 * 64 * 320), 256, 0, stream>>>(word_rel,  word_emb, 300, 12, xwc, 12 * 64 * 320);
  k_gather<<<nb(4  * 64 * 320), 256, 0, stream>>>(rela_rel,  rela_emb, 300, 4,  xrc, 4 * 64 * 320);

  // ---- BiLSTMs (out layout (B,T,512): fwd at +0, bwd at +256, bwd time-reversed) ----
  k_lstm<<<1, 512, 0, stream>>>(xq, 40, 0, wihF, whhF, b_f, nullptr, nullptr, hstF, cstF, qout, 0,   40 * 512);
  k_lstm<<<1, 512, 0, stream>>>(xq, 40, 1, wihB, whhB, b_b, nullptr, nullptr, hstB, cstB, qout, 256, 40 * 512);
  // previous hop: word then rela (state carry per direction)
  k_lstm<<<1, 512, 0, stream>>>(xwp, 12, 0, wihF, whhF, b_f, nullptr, nullptr, hstF, cstF, wpo, 0,   12 * 512);
  k_lstm<<<1, 512, 0, stream>>>(xrp, 4,  0, wihF, whhF, b_f, hstF, cstF, hstD, cstD, rpo, 0,   4 * 512);
  k_lstm<<<1, 512, 0, stream>>>(xwp, 12, 1, wihB, whhB, b_b, nullptr, nullptr, hstB, cstB, wpo, 256, 12 * 512);
  k_lstm<<<1, 512, 0, stream>>>(xrp, 4,  1, wihB, whhB, b_b, hstB, cstB, hstD, cstD, rpo, 256, 4 * 512);
  // current relation
  k_lstm<<<1, 512, 0, stream>>>(xwc, 12, 0, wihF, whhF, b_f, nullptr, nullptr, hstF, cstF, wco, 0,   12 * 512);
  k_lstm<<<1, 512, 0, stream>>>(xrc, 4,  0, wihF, whhF, b_f, hstF, cstF, hstD, cstD, rco, 0,   4 * 512);
  k_lstm<<<1, 512, 0, stream>>>(xwc, 12, 1, wihB, whhB, b_b, nullptr, nullptr, hstB, cstB, wco, 256, 12 * 512);
  k_lstm<<<1, 512, 0, stream>>>(xrc, 4,  1, wihB, whhB, b_b, hstB, cstB, hstD, cstD, rco, 256, 4 * 512);

  // ---- relation buffers ----
  k_relation<<<nb(64 * 16 * 512), 256, 0, stream>>>(rpo, wpo, relP);
  k_relation<<<nb(64 * 16 * 512), 256, 0, stream>>>(rco, wco, relC);

  // ---- previous-hop attention, subtract, lin2 ----
  k_attention<<<64, 512, 0, stream>>>(relP, qout, WT, attP);
  k_sub_f16<<<nb(64 * 40 * 512), 256, 0, stream>>>(qout, attP, subT, 64 * 40 * 512);
  k_gemm_tn<<<(160 * 32) / 8, 256, 0, stream>>>(subT, 512, lin2h, 512, 512, lin2_b,
                                                qt2, 512, 160, 32);

  // ---- current attention, concat, convs + score ----
  k_attention<<<64, 512, 0, stream>>>(relC, qt2, WT, attC);
  k_concat_M<<<nb(64 * 40 * 1024), 256, 0, stream>>>(qt2, attC, Mbuf, 64 * 40 * 1024);
  k_conv_score<<<64, 512, 0, stream>>>(Mbuf, wr1, wr2, wr3, cnn1_b, cnn2_b, cnn3_b,
                                       lin_w, outp);
}
